// Gemma2Attention_29008209117626
// MI455X (gfx1250) — matrix operand order repaired and checked by reading, not compiled, not run on hardware
//
#include <hip/hip_runtime.h>
#include <hip/hip_bf16.h>

typedef __bf16 bf16;
typedef __attribute__((ext_vector_type(16))) __bf16    v16bf;
typedef __attribute__((ext_vector_type(8)))  float     v8f;
typedef __attribute__((ext_vector_type(4)))  unsigned  v4u;
typedef __attribute__((ext_vector_type(8)))  int       v8i;
typedef __attribute__((ext_vector_type(4)))  int       v4i;

#define HID   3584
#define NH    16
#define NKV   8
#define HD    256
#define SEQ   2048
#define QKV_N ((NH + 2 * NKV) * HD)   /* 8192 */
#define ATT_N (NH * HD)               /* 4096 */

__constant__ const float ATT_SCALE = 0.0625f;   /* 256^-0.5 */
__constant__ const float CAP       = 50.0f;

// ---------------------------------------------------------------------------
// WMMA helpers (wave32, 16x16x32 bf16 -> f32)
// ---------------------------------------------------------------------------
__device__ inline v8f wmma_bf16(v16bf a, v16bf b, v8f c) {
    return __builtin_amdgcn_wmma_f32_16x16x32_bf16(
        false, a, false, b, (short)0, c, false, false);
}

// A fragment: 16x32 bf16 tile, row-major [m][k], `stride` elems/row.
__device__ inline v16bf load_a_frag(const bf16* p0, int stride) {
    const int lane = threadIdx.x & 31;
    const int hi   = lane >> 4;
    const bf16* p  = p0 + (lane & 15) * stride + hi * 8;
    union { v16bf v; unsigned u[8]; } r;
#pragma unroll
    for (int i = 0; i < 4; ++i) r.u[i]     = *(const unsigned*)(p + i * 2);
#pragma unroll
    for (int i = 0; i < 4; ++i) r.u[4 + i] = *(const unsigned*)(p + 16 + i * 2);
    return r.v;
}

// B fragment: 32x16 tile staged K-transposed in LDS: [n][k], `stride` elems/row.
__device__ inline v16bf load_b_frag(const bf16* p0, int stride) {
    const int lane = threadIdx.x & 31;
    const bf16* p  = p0 + (lane & 15) * stride + (lane >> 4) * 8;
    union { v16bf v; unsigned u[8]; } r;
#pragma unroll
    for (int i = 0; i < 4; ++i) r.u[i]     = *(const unsigned*)(p + i * 2);
#pragma unroll
    for (int i = 0; i < 4; ++i) r.u[4 + i] = *(const unsigned*)(p + 16 + i * 2);
    return r.v;
}

// Async global->LDS 16-byte copy (ASYNCcnt). ldsoff = LDS byte address.
__device__ inline void async_copy_b128(unsigned ldsoff, const void* gsrc) {
    unsigned long long ga = (unsigned long long)(size_t)gsrc;
    asm volatile("global_load_async_to_lds_b128 %0, %1, off"
                 :: "v"(ldsoff), "v"(ga) : "memory");
}
__device__ inline void wait_async0() {
    asm volatile("s_wait_asynccnt 0x0" ::: "memory");
}

// ---------------------------------------------------------------------------
// C[M,N] = A[M,K] * B[K,N]; A f32 or bf16, B f32, C f32.
// Block 128x128, 8 waves (4M x 2N); each wave 32x64 = 2x4 WMMA tiles.
// ---------------------------------------------------------------------------
template <typename AT>
__global__ __launch_bounds__(256)
void gemm_bf16(const AT* __restrict__ A, const float* __restrict__ B,
               float* __restrict__ C, int M, int N, int K)
{
    constexpr int AS = 34;
    constexpr int BS = 34;
    __shared__ bf16 As[128 * AS];
    __shared__ bf16 Bs[128 * BS];

    const int tid  = threadIdx.x;
    const int lane = tid & 31;
    const int wave = tid >> 5;
    const int bm   = blockIdx.y * 128;
    const int bn   = blockIdx.x * 128;
    const int wm   = (wave >> 1) * 32;
    const int wn   = (wave & 1) * 64;

    v8f acc[2][4] = {};

    for (int k0 = 0; k0 < K; k0 += 32) {
        // Stage A tile 128x32 (cvt -> bf16)
#pragma unroll 4
        for (int i = 0; i < 16; ++i) {
            int idx = i * 256 + tid;
            int m = idx >> 5, kk = idx & 31;
            As[m * AS + kk] = (bf16)(float)A[(size_t)(bm + m) * K + (k0 + kk)];
        }
        // Stage B tile 32x128 transposed -> [n][k]
#pragma unroll 4
        for (int i = 0; i < 16; ++i) {
            int idx = i * 256 + tid;
            int kk = idx >> 7, n = idx & 127;
            Bs[n * BS + kk] = (bf16)B[(size_t)(k0 + kk) * N + (bn + n)];
        }
        __syncthreads();

        v16bf af0 = load_a_frag(As + (wm +  0) * AS, AS);
        v16bf af1 = load_a_frag(As + (wm + 16) * AS, AS);
#pragma unroll
        for (int ni = 0; ni < 4; ++ni) {
            v16bf bfr = load_b_frag(Bs + (wn + ni * 16) * BS, BS);
            acc[0][ni] = wmma_bf16(af0, bfr, acc[0][ni]);
            acc[1][ni] = wmma_bf16(af1, bfr, acc[1][ni]);
        }
        __syncthreads();
    }

    const int n  = lane & 15;
    const int hi = lane >> 4;
#pragma unroll
    for (int mi = 0; mi < 2; ++mi)
#pragma unroll
        for (int ni = 0; ni < 4; ++ni)
#pragma unroll
            for (int r = 0; r < 8; ++r) {
                int mm = bm + wm + mi * 16 + r + hi * 8;
                int nn = bn + wn + ni * 16 + n;
                C[(size_t)mm * N + nn] = acc[mi][ni][r];
            }
}

// ---------------------------------------------------------------------------
// RoPE (neox, full head dim) + split qkv f32 -> q/k/v bf16
// ---------------------------------------------------------------------------
__global__ __launch_bounds__(256)
void rope_split_kernel(const float* __restrict__ qkv, const int* __restrict__ positions,
                       bf16* __restrict__ qg, bf16* __restrict__ kg, bf16* __restrict__ vg)
{
    int idx  = blockIdx.x * blockDim.x + threadIdx.x;
    int i    = idx & 127;
    int slot = (idx >> 7) & 31;
    int t    = idx >> 12;
    const float* row = qkv + (size_t)t * QKV_N + slot * HD;

    if (slot < NH + NKV) {
        float inv = __powf(10000.0f, -(float)(2 * i) * (1.0f / 256.0f));
        float ang = (float)positions[t] * inv;
        float s, c;
        __sincosf(ang, &s, &c);
        float x1 = row[i], x2 = row[i + 128];
        float o1 = x1 * c - x2 * s;
        float o2 = x2 * c + x1 * s;
        if (slot < NH) {
            bf16* dst = qg + ((size_t)t * NH + slot) * HD;
            dst[i] = (bf16)o1; dst[i + 128] = (bf16)o2;
        } else {
            bf16* dst = kg + ((size_t)t * NKV + (slot - NH)) * HD;
            dst[i] = (bf16)o1; dst[i + 128] = (bf16)o2;
        }
    } else {
        bf16* dst = vg + ((size_t)t * NKV + (slot - NH - NKV)) * HD;
        dst[i] = (bf16)row[i]; dst[i + 128] = (bf16)row[i + 128];
    }
}

// ---------------------------------------------------------------------------
// Flash attention: 4 waves/WG, 16 q/wave (64 q/block), 32-key blocks.
// K tile via TDM tensor_load_to_lds (padded layout: 512B rows + 16B pad ->
// 528B stride == KSTR). V tile via global_load_async_to_lds_b128, then
// LDS->LDS transpose for the P*V B-fragments.
// ---------------------------------------------------------------------------
__global__ __launch_bounds__(128)
void flash_attn_kernel(const bf16* __restrict__ qg, const bf16* __restrict__ kg,
                       const bf16* __restrict__ vg, bf16* __restrict__ outg)
{
    constexpr int KSTR = HD + 8;   // 264 elems = 528 B rows
    constexpr int VSTR = 34;
    __shared__ bf16 Ks  [32 * KSTR];     // [key][d]
    __shared__ bf16 Vraw[32 * KSTR];     // [key][d] (async-copied)
    __shared__ bf16 Vt  [HD * VSTR];     // [d][key]
    __shared__ bf16 Ps  [4][16 * 34];

    const int tid   = threadIdx.x;
    const int lane  = tid & 31;
    const int wave  = tid >> 5;
    const int h     = blockIdx.x % NH;
    const int qblk  = blockIdx.x / NH;
    const int qBase = qblk * 64;
    const int qRow0 = qBase + wave * 16;
    const int kvh   = h >> 1;            // GQA rep = 2
    const int hi    = lane >> 4;
    const int n     = lane & 15;

    // Preload Q fragments (16 rows x 256 d) from global
    v16bf qf[8];
    {
        const bf16* qp = qg + ((size_t)(qRow0 + n) * NH + h) * HD;
#pragma unroll
        for (int dc = 0; dc < 8; ++dc) {
            union { v16bf v; unsigned u[8]; } r;
            const bf16* p = qp + dc * 32 + hi * 8;
#pragma unroll
            for (int i = 0; i < 4; ++i) r.u[i]     = *(const unsigned*)(p + i * 2);
#pragma unroll
            for (int i = 0; i < 4; ++i) r.u[4 + i] = *(const unsigned*)(p + 16 + i * 2);
            qf[dc] = r.v;
        }
    }

    float rowMax[8], rowSum[8];
#pragma unroll
    for (int r = 0; r < 8; ++r) { rowMax[r] = -1e30f; rowSum[r] = 0.0f; }
    v8f o[16] = {};

    const int nkb = qBase / 32 + 2;
    for (int kblk = 0; kblk < nkb; ++kblk) {
        const int kStart = kblk * 32;

        // ---- K tile: Tensor Data Mover, one descriptor per workgroup ----
#if __has_builtin(__builtin_amdgcn_tensor_load_to_lds)
        if (wave == 0) {
            const bf16* gsrc = kg + ((size_t)kStart * NKV + kvh) * HD;
            unsigned long long ga = (unsigned long long)(size_t)gsrc;
            v4u g0;
            g0.x = 1u;                                        // count=1
            g0.y = (unsigned)(size_t)(const void*)&Ks[0];     // lds_addr
            g0.z = (unsigned)(ga & 0xffffffffu);              // global_addr lo
            g0.w = (unsigned)((ga >> 32) & 0x01ffffffu) | (2u << 30); // hi + type=2
            v8i g1;
            g1[0] = (int)((1u << 16)      // data_size = 2B
                        | (1u << 20)      // pad_enable
                        | (6u << 22)      // pad_interval = 128 DW (512B)
                        | (3u << 25));    // pad_amount   = 4 DW (16B)
            g1[1] = (int)(256u << 16);    // tensor_dim0 = 256
            g1[2] = (int)(32u  << 16);    // tensor_dim1 = 32
            g1[3] = (int)(256u << 16);    // tile_dim0   = 256
            g1[4] = 32;                   // tile_dim1   = 32
            g1[5] = NKV * HD;             // tensor_dim0_stride = 2048
            g1[6] = 0;
            g1[7] = 0;
            v4i z4 = {0, 0, 0, 0};
            v8i z8 = {0, 0, 0, 0, 0, 0, 0, 0};
            __builtin_amdgcn_tensor_load_to_lds(g0, g1, z4, z4, z8, 0);
        }
#else
        for (int i = 0; i < 8; ++i) {
            int c = i * 128 + tid;                   // 1024 x 16B chunks
            int kk = c >> 5, d0 = (c & 31) * 8;
            async_copy_b128((unsigned)(size_t)(const void*)(Ks + kk * KSTR + d0),
                            kg + ((size_t)(kStart + kk) * NKV + kvh) * HD + d0);
        }
#endif
        // ---- V tile: async global->LDS (verbatim), transpose afterwards ----
        for (int i = 0; i < 8; ++i) {
            int c = i * 128 + tid;
            int kk = c >> 5, d0 = (c & 31) * 8;
            async_copy_b128((unsigned)(size_t)(const void*)(Vraw + kk * KSTR + d0),
                            vg + ((size_t)(kStart + kk) * NKV + kvh) * HD + d0);
        }
        wait_async0();
#if __has_builtin(__builtin_amdgcn_s_wait_tensorcnt)
        __builtin_amdgcn_s_wait_tensorcnt(0);
#endif
        __syncthreads();

        // LDS->LDS transpose: Vraw[key][d] -> Vt[d][key]
        for (int i = 0; i < 64; ++i) {
            int idx = i * 128 + tid;
            int kk = idx >> 8, d = idx & 255;
            Vt[d * VSTR + kk] = Vraw[kk * KSTR + d];
        }
        __syncthreads();

        // S = Q K^T (two 16-key tiles)
        v8f s0 = {}, s1 = {};
#pragma unroll
        for (int dc = 0; dc < 8; ++dc) {
            v16bf b0 = load_b_frag(Ks + dc * 32, KSTR);
            s0 = wmma_bf16(qf[dc], b0, s0);
            v16bf b1 = load_b_frag(Ks + 16 * KSTR + dc * 32, KSTR);
            s1 = wmma_bf16(qf[dc], b1, s1);
        }

        // soft-cap, causal mask, online softmax
        float pr0[8], pr1[8];
#pragma unroll
        for (int r = 0; r < 8; ++r) {
            int qi = qRow0 + r + hi * 8;
            float l0 = CAP * tanhf(s0[r] * ATT_SCALE * (1.0f / CAP));
            float l1 = CAP * tanhf(s1[r] * ATT_SCALE * (1.0f / CAP));
            if (kStart + n > qi)      l0 = -1e30f;
            if (kStart + 16 + n > qi) l1 = -1e30f;
            float tmax = fmaxf(l0, l1);
#pragma unroll
            for (int m = 8; m >= 1; m >>= 1) tmax = fmaxf(tmax, __shfl_xor(tmax, m, 32));
            float nm = fmaxf(rowMax[r], tmax);
            float sc = __expf(rowMax[r] - nm);
            float p0 = (l0 <= -1e29f) ? 0.0f : __expf(l0 - nm);
            float p1 = (l1 <= -1e29f) ? 0.0f : __expf(l1 - nm);
            float ps = p0 + p1;
#pragma unroll
            for (int m = 8; m >= 1; m >>= 1) ps += __shfl_xor(ps, m, 32);
            rowSum[r] = rowSum[r] * sc + ps;
            rowMax[r] = nm;
#pragma unroll
            for (int dt = 0; dt < 16; ++dt) o[dt][r] *= sc;
            pr0[r] = p0; pr1[r] = p1;
        }

        // P (C-layout) -> per-wave LDS -> A fragment; O += P V
        bf16* pw = &Ps[wave][0];
#pragma unroll
        for (int r = 0; r < 8; ++r) {
            int m = r + hi * 8;
            pw[m * 34 + n]      = (bf16)pr0[r];
            pw[m * 34 + 16 + n] = (bf16)pr1[r];
        }
        v16bf pa = load_a_frag(pw, 34);
#pragma unroll
        for (int dt = 0; dt < 16; ++dt) {
            v16bf vb = load_b_frag(Vt + dt * 16 * VSTR, VSTR);
            o[dt] = wmma_bf16(pa, vb, o[dt]);
        }
        __syncthreads();
    }

    // normalize + write bf16
#pragma unroll
    for (int r = 0; r < 8; ++r) {
        float inv = 1.0f / rowSum[r];
        int qi = qRow0 + r + hi * 8;
        bf16* op = outg + (size_t)qi * ATT_N + h * HD;
#pragma unroll
        for (int dt = 0; dt < 16; ++dt)
            op[dt * 16 + n] = (bf16)(o[dt][r] * inv);
    }
}

// ---------------------------------------------------------------------------
extern "C" void kernel_launch(void* const* d_in, const int* in_sizes, int n_in,
                              void* d_out, int out_size, void* d_ws, size_t ws_size,
                              hipStream_t stream)
{
    const int*   positions = (const int*)  d_in[0];
    const float* hidden    = (const float*)d_in[1];
    const float* w_qkv     = (const float*)d_in[2];
    const float* w_o       = (const float*)d_in[3];
    float* out = (float*)d_out;

    char* ws = (char*)d_ws;
    float* qkv  = (float*)ws;
    bf16*  qb   = (bf16*)(ws + (size_t)SEQ * QKV_N * 4);
    bf16*  kb   = (bf16*)((char*)qb + (size_t)SEQ * NH  * HD * 2);
    bf16*  vb   = (bf16*)((char*)kb + (size_t)SEQ * NKV * HD * 2);
    bf16*  attn = (bf16*)((char*)vb + (size_t)SEQ * NKV * HD * 2);

    dim3 g1(QKV_N / 128, SEQ / 128);
    gemm_bf16<float><<<g1, 256, 0, stream>>>(hidden, w_qkv, qkv, SEQ, QKV_N, HID);

    int total = SEQ * 32 * 128;
    rope_split_kernel<<<total / 256, 256, 0, stream>>>(qkv, positions, qb, kb, vb);

    flash_attn_kernel<<<NH * (SEQ / 64), 128, 0, stream>>>(qb, kb, vb, attn);

    dim3 g2(HID / 128, SEQ / 128);
    gemm_bf16<bf16><<<g2, 256, 0, stream>>>(attn, w_o, out, SEQ, HID, ATT_N);
}
